// SplitResidualVectorQuantizer_69913477644918
// MI455X (gfx1250) — compile-verified, matrix operand order checked by reading, not run on hardware
//
#include <hip/hip_runtime.h>
#include <hip/hip_bf16.h>

#define NQ   8
#define CBK  2048
#define DIM  512
#define BB   32
#define TT   1500
#define NN   (BB * TT)          // 48000 rows
#define QELEMS ((size_t)BB * DIM * TT)
#define IELEMS ((size_t)BB * NQ * TT)

typedef __attribute__((ext_vector_type(16))) __bf16 v16bf;
typedef __attribute__((ext_vector_type(8)))  __bf16 v8bf;
typedef __attribute__((ext_vector_type(8)))  float  v8f;

// ---------------------------------------------------------------------------
// Prep: codebook -> bf16 copy, and per-row squared L2 norms. One wave per row.
// ---------------------------------------------------------------------------
__global__ __launch_bounds__(256)
void rvq_prep_kernel(const float* __restrict__ cb_f32,
                     __bf16* __restrict__ cb_bf16,
                     float* __restrict__ cbnorm) {
    const int wave = (blockIdx.x * (blockDim.x >> 5)) + (threadIdx.x >> 5);
    const int lane = threadIdx.x & 31;
    if (wave >= NQ * CBK) return;
    const float* row = cb_f32 + (size_t)wave * DIM;
    __bf16* brow = cb_bf16 + (size_t)wave * DIM;
    float s = 0.0f;
    #pragma unroll
    for (int d = lane; d < DIM; d += 32) {
        float v = row[d];
        s += v * v;
        brow[d] = (__bf16)v;
    }
    #pragma unroll
    for (int m = 16; m >= 1; m >>= 1) s += __shfl_xor(s, m, 32);
    if (lane == 0) cbnorm[wave] = s;
}

// ---------------------------------------------------------------------------
// x [B, D, T] -> residual [B*T, D] row-major (coalesced writes along D)
// ---------------------------------------------------------------------------
__global__ __launch_bounds__(256)
void rvq_transpose_in_kernel(const float* __restrict__ x,
                             float* __restrict__ residual) {
    size_t i = (size_t)blockIdx.x * blockDim.x + threadIdx.x;
    if (i >= (size_t)NN * DIM) return;
    int d = (int)(i & (DIM - 1));
    size_t nt = i >> 9;                 // / DIM
    int t = (int)(nt % TT);
    int b = (int)(nt / TT);
    residual[i] = x[((size_t)b * DIM + d) * TT + t];
}

// ---------------------------------------------------------------------------
// Argmin over 2048 codes via WMMA bf16. 4 waves/block, 16 rows/wave.
// A-tile (16x512) held in VGPRs as 16 bf16 fragments (K=32 each).
// ---------------------------------------------------------------------------
__global__ __launch_bounds__(128)
void rvq_argmin_kernel(const float* __restrict__ residual,
                       const __bf16* __restrict__ cb,      // [CBK, DIM] bf16
                       const float* __restrict__ cbnorm,   // [CBK]
                       int* __restrict__ idx_out) {
    const int lane = threadIdx.x & 31;
    const int wave = threadIdx.x >> 5;
    const int rowbase = (blockIdx.x * 4 + wave) * 16;
    const int l15 = lane & 15;
    const int klo = (lane >> 4) * 8;    // K sub-offset per lane half

    // Load + convert the A tile: 16 fragments of 16x32 bf16
    v16bf a[16];
    {
        const float* rptr = residual + (size_t)(rowbase + l15) * DIM + klo;
        #pragma unroll
        for (int ks = 0; ks < 16; ++ks) {
            const float* p0 = rptr + ks * 32;
            const float* p1 = p0 + 16;
            v16bf f;
            #pragma unroll
            for (int j = 0; j < 8; ++j) {
                f[j]     = (__bf16)p0[j];
                f[j + 8] = (__bf16)p1[j];
            }
            a[ks] = f;
        }
    }

    float dmin[8];
    int   imin[8];
    #pragma unroll
    for (int r = 0; r < 8; ++r) { dmin[r] = 3.0e38f; imin[r] = 0; }

    for (int cbase = 0; cbase < CBK; cbase += 16) {
        const int code = cbase + l15;                 // this lane's code column
        const __bf16* cptr = cb + (size_t)code * DIM + klo;
        v8f acc = {};
        #pragma unroll
        for (int ks = 0; ks < 16; ++ks) {
            v8bf b0 = *(const v8bf*)(cptr + ks * 32);
            v8bf b1 = *(const v8bf*)(cptr + ks * 32 + 16);
            v16bf bb;
            #pragma unroll
            for (int j = 0; j < 8; ++j) { bb[j] = b0[j]; bb[j + 8] = b1[j]; }
            acc = __builtin_amdgcn_wmma_f32_16x16x32_bf16(
                      false, a[ks], false, bb, (short)0, acc, false, false);
        }
        const float nrm = cbnorm[code];
        #pragma unroll
        for (int r = 0; r < 8; ++r) {
            float dist = nrm - 2.0f * acc[r];         // ||x||^2 constant: drop
            if (dist < dmin[r]) { dmin[r] = dist; imin[r] = code; }
        }
    }

    // Cross-lane argmin within each 16-lane half (tiebreak: lowest index)
    #pragma unroll
    for (int r = 0; r < 8; ++r) {
        float v = dmin[r];
        int   id = imin[r];
        #pragma unroll
        for (int m = 8; m >= 1; m >>= 1) {
            float ov  = __shfl_xor(v, m, 16);
            int   oid = __shfl_xor(id, m, 16);
            if (ov < v || (ov == v && oid < id)) { v = ov; id = oid; }
        }
        if (l15 == 0) idx_out[rowbase + r + klo] = id;  // klo = 8*half -> row
    }
}

// ---------------------------------------------------------------------------
// Residual update + loss + index emission. One block per row n.
// ---------------------------------------------------------------------------
__global__ __launch_bounds__(128)
void rvq_update_kernel(const int* __restrict__ idx,
                       const float* __restrict__ cb_f32,  // stage slice [CBK, DIM]
                       float* __restrict__ residual,
                       float* __restrict__ out_idx,        // d_out + QELEMS
                       float* __restrict__ loss_acc,
                       int q) {
    const int n = blockIdx.x;
    const int tid = threadIdx.x;
    const int id = idx[n];
    const float* crow = cb_f32 + (size_t)id * DIM;
    float* rrow = residual + (size_t)n * DIM;

    float local = 0.0f;
    #pragma unroll
    for (int d = tid; d < DIM; d += 128) {
        float cq = crow[d];
        float r  = rrow[d];
        float diff = cq - r;
        local += diff * diff;
        rrow[d] = -diff;                 // residual - quant
    }

    __shared__ float red[128];
    red[tid] = local;
    __syncthreads();
    #pragma unroll
    for (int s = 64; s > 0; s >>= 1) {
        if (tid < s) red[tid] += red[tid + s];
        __syncthreads();
    }
    if (tid == 0) {
        atomicAdd(loss_acc, red[0]);
        int b = n / TT, t = n % TT;
        out_idx[(size_t)b * (NQ * TT) + (size_t)q * TT + t] = (float)id;
    }
}

// ---------------------------------------------------------------------------
// quantized_out = x - residual_final (telescoping sum), transposed back to
// [B, D, T] with an LDS-tiled transpose (coalesced on both sides).
// ---------------------------------------------------------------------------
__global__ __launch_bounds__(256)
void rvq_output_kernel(const float* __restrict__ x,
                       const float* __restrict__ residual,
                       float* __restrict__ out) {
    __shared__ float tile[32][33];
    const int b  = blockIdx.z;
    const int t0 = blockIdx.x * 32;
    const int d0 = blockIdx.y * 32;
    const int tx = threadIdx.x;          // 32
    const int ty = threadIdx.y;          // 8

    #pragma unroll
    for (int i = ty; i < 32; i += 8) {
        int t = t0 + i;
        if (t < TT)
            tile[i][tx] = residual[((size_t)b * TT + t) * DIM + d0 + tx];
    }
    __syncthreads();
    #pragma unroll
    for (int i = ty; i < 32; i += 8) {
        int t = t0 + tx;
        if (t < TT) {
            size_t o = ((size_t)b * DIM + d0 + i) * TT + t;
            out[o] = x[o] - tile[tx][i];
        }
    }
}

__global__ void rvq_finalize_kernel(const float* __restrict__ loss_acc,
                                    float* __restrict__ out) {
    out[QELEMS + IELEMS] = loss_acc[0] / (float)((size_t)BB * DIM * TT);
}

// ---------------------------------------------------------------------------
extern "C" void kernel_launch(void* const* d_in, const int* in_sizes, int n_in,
                              void* d_out, int out_size, void* d_ws, size_t ws_size,
                              hipStream_t stream) {
    const float* x         = (const float*)d_in[0];   // [B, D, T]
    const float* codebooks = (const float*)d_in[1];   // [NQ, CBK, DIM]
    float* out = (float*)d_out;

    // Workspace carve-up (~115 MB)
    char* ws = (char*)d_ws;
    size_t off = 0;
    float*  residual = (float*)(ws + off);  off += (size_t)NN * DIM * sizeof(float);
    __bf16* cb_bf16  = (__bf16*)(ws + off); off += (size_t)NQ * CBK * DIM * sizeof(__bf16);
    float*  cbnorm   = (float*)(ws + off);  off += (size_t)NQ * CBK * sizeof(float);
    int*    idxbuf   = (int*)(ws + off);    off += (size_t)NN * sizeof(int);
    float*  loss_acc = (float*)(ws + off);  off += 256;

    hipMemsetAsync(loss_acc, 0, sizeof(float), stream);

    // Codebook bf16 + norms (8 waves per 256-thread block)
    rvq_prep_kernel<<<(NQ * CBK) / 8, 256, 0, stream>>>(codebooks, cb_bf16, cbnorm);

    // x -> residual [N, D]
    rvq_transpose_in_kernel<<<(NN * DIM) / 256, 256, 0, stream>>>(x, residual);

    for (int q = 0; q < NQ; ++q) {
        rvq_argmin_kernel<<<NN / 64, 128, 0, stream>>>(
            residual,
            cb_bf16 + (size_t)q * CBK * DIM,
            cbnorm + (size_t)q * CBK,
            idxbuf);
        rvq_update_kernel<<<NN, 128, 0, stream>>>(
            idxbuf,
            codebooks + (size_t)q * CBK * DIM,
            residual,
            out + QELEMS,
            loss_acc,
            q);
    }

    dim3 tgrid((TT + 31) / 32, DIM / 32, BB);
    rvq_output_kernel<<<tgrid, dim3(32, 8, 1), 0, stream>>>(x, residual, out);
    rvq_finalize_kernel<<<1, 1, 0, stream>>>(loss_acc, out);
}
